// DownsampleGRVQ_65128884076852
// MI455X (gfx1250) — compile-verified, matrix operand order checked by reading, not run on hardware
//
#include <hip/hip_runtime.h>
#include <hip/hip_bf16.h>

typedef unsigned short bf16_t;
typedef __attribute__((ext_vector_type(16))) __bf16 v16bf;
typedef __attribute__((ext_vector_type(8)))  float  v8f;

static __device__ __forceinline__ bf16_t f2bf(float f) {
  unsigned u = __float_as_uint(f);
  unsigned r = (u + 0x7FFFu + ((u >> 16) & 1u)) >> 16;  // round-to-nearest-even
  return (bf16_t)r;
}

union FragU { unsigned u[8]; v16bf v; };

#define EPI_BIAS 0
#define EPI_GELU 1
#define EPI_RES  2

// CDNA5 async global->LDS copy (ASYNCcnt-tracked). vdst = per-lane LDS byte
// offset, vaddr = per-lane 64-bit global address.
static __device__ __forceinline__ void async_ld_b128(unsigned lds_off, const void* gptr) {
  asm volatile("global_load_async_to_lds_b128 %0, %1, off"
               :: "v"(lds_off), "v"((unsigned long long)(uintptr_t)gptr)
               : "memory");
}
static __device__ __forceinline__ void wait_async0() {
  asm volatile("s_wait_asynccnt 0" ::: "memory");
}

// ---------------------------------------------------------------------------
// Conversion / packing kernels
// ---------------------------------------------------------------------------
__global__ void k_f32_to_bf16(const float* __restrict__ s, bf16_t* __restrict__ d, int n) {
  int i = blockIdx.x * 256 + threadIdx.x;
  if (i < n) d[i] = f2bf(s[i]);
}

// Build B for stride-2 down conv: Bc[(i*2+k)][t] = x[i][2t+k], x is (512,T) f32
__global__ void k_pack_downconv_B(const float* __restrict__ x, bf16_t* __restrict__ Bc, int T) {
  int Th = T >> 1;
  int i = blockIdx.x * 256 + threadIdx.x;
  int total = 1024 * Th;
  if (i >= total) return;
  int row = i / Th, t = i - row * Th;
  int ch = row >> 1, k = row & 1;
  Bc[i] = f2bf(x[(size_t)ch * T + 2 * t + k]);
}

// Up-conv weights (I=512,O=512,K=2): A_k[o][i] = w[i][o][k] (bf16, row-major)
__global__ void k_pack_upconv_A(const float* __restrict__ w, bf16_t* __restrict__ A0,
                                bf16_t* __restrict__ A1) {
  int i = blockIdx.x * 256 + threadIdx.x;  // o*512 + ci
  if (i >= 512 * 512) return;
  int o = i >> 9, ci = i & 511;
  A0[i] = f2bf(w[(size_t)ci * 1024 + o * 2 + 0]);
  A1[i] = f2bf(w[(size_t)ci * 1024 + o * 2 + 1]);
}

__global__ void k_cb_norms(const float* __restrict__ cb, float* __restrict__ n2) {
  int j = blockIdx.x * 256 + threadIdx.x;
  if (j >= 1024) return;
  const float* r = cb + (size_t)j * 512;
  float s = 0.f;
  for (int c = 0; c < 512; ++c) s += r[c] * r[c];
  n2[j] = s;
}

__global__ void k_zero(float* p) { *p = 0.f; }

// ---------------------------------------------------------------------------
// WMMA GEMM: C[M,N] = A[M,K] * B[K,N], A/B bf16 row-major, f32 accumulate.
// Block = 256 threads (8 waves, wave32). Block tile 128x64.
// Each wave owns a 16x64 strip: 1 A fragment reused by 4 B fragments
// -> 4 v_wmma_f32_16x16x32_bf16 per K-step.
// Double-buffered LDS pipeline: tile i+1 is staged (A via
// global_load_async_to_lds_b128, B via b128 load issued up-front and scattered
// transposed AFTER compute) while the 4 WMMAs run on tile i. One barrier and
// one s_wait_asynccnt per K-step; global_prefetch_b8 pulls tile i+2.
// Epilogues: bias / bias+GELU(exact) / residual + gamma*(acc+bias).
// nstride=2 interleaves columns (transposed-conv store).
// ---------------------------------------------------------------------------
#define APITCH 40  // half-words; 80B row stride keeps b128 LDS writes 16B-aligned
#define BPITCH 34  // half-words; 68B row stride keeps b32 frag reads 4B-aligned

__global__ __launch_bounds__(256) void k_gemm_wmma(
    const bf16_t* __restrict__ A, const bf16_t* __restrict__ B,
    float* __restrict__ C, int N, int K, int ldc, int nstride,
    const float* __restrict__ bias, const float* __restrict__ gamma,
    const float* __restrict__ resid, int epi) {
  __shared__ bf16_t As[2][128 * APITCH];  // [m][k], double-buffered
  __shared__ bf16_t Bs[2][64 * BPITCH];   // transposed [n][k], double-buffered

  const int tid  = threadIdx.x;
  const int lane = tid & 31;
  const int wave = tid >> 5;           // 0..7 -> M sub-tile
  const int m0 = blockIdx.y * 128;
  const int n0 = blockIdx.x * 64;

  v8f acc[4];
#pragma unroll
  for (int nt = 0; nt < 4; ++nt) acc[nt] = (v8f)0.0f;

  const int ar = tid >> 1, ac = (tid & 1) * 16;  // A staging: 128 rows x 32 cols
  const int br = tid >> 3, bc = (tid & 7) * 8;   // B staging: 32 rows x 64 cols
  unsigned abase[2];
  abase[0] = (unsigned)(uintptr_t)(void*)&As[0][0];
  abase[1] = (unsigned)(uintptr_t)(void*)&As[1][0];
  const unsigned aoff = (unsigned)(ar * APITCH + ac) * 2u;
  const bf16_t* gA = A + (size_t)(m0 + ar) * K + ac;

  const int nk = K >> 5;

  // ---- prologue: stage tile 0 into buffer 0 ----
  {
    async_ld_b128(abase[0] + aoff, gA);
    async_ld_b128(abase[0] + aoff + 16u, gA + 8);
    union { uint4 q; bf16_t s[8]; } u;
    u.q = *(const uint4*)(B + (size_t)br * N + n0 + bc);
#pragma unroll
    for (int e = 0; e < 8; ++e) Bs[0][(bc + e) * BPITCH + br] = u.s[e];
  }
  wait_async0();
  __syncthreads();

  for (int i = 0; i < nk; ++i) {
    const int cur = i & 1;
    const int nxt = cur ^ 1;
    const int k0 = i << 5;
    const bool have_next = (i + 1 < nk);

    // Kick off tile i+1: async A copies + B global load (consumed after compute)
    union { uint4 q; bf16_t s[8]; } ub;
    if (have_next) {
      const bf16_t* ga = gA + k0 + 32;
      async_ld_b128(abase[nxt] + aoff, ga);
      async_ld_b128(abase[nxt] + aoff + 16u, ga + 8);
      ub.q = *(const uint4*)(B + (size_t)(k0 + 32 + br) * N + n0 + bc);
      if (i + 2 < nk)
        __builtin_prefetch(B + (size_t)(k0 + 64 + br) * N + n0 + bc, 0, 3);
    }

    // Compute on tile i.
    // Fragments per ISA 16-bit layouts: lanes 0-15 K-base 0, lanes 16-31 K-base 8;
    // VGPR v<4 covers K in [kb,kb+8), v>=4 covers [16+kb,16+kb+8), pairs packed.
    const int kb = (lane >> 4) << 3;
    FragU fa;
    const bf16_t* arow = &As[cur][0] + (wave * 16 + (lane & 15)) * APITCH;
#pragma unroll
    for (int v = 0; v < 4; ++v) {
      fa.u[v]     = *(const unsigned*)(arow + kb + 2 * v);
      fa.u[4 + v] = *(const unsigned*)(arow + 16 + kb + 2 * v);
    }
    FragU fb[4];
#pragma unroll
    for (int nt = 0; nt < 4; ++nt) {
      const bf16_t* brow = &Bs[cur][0] + (nt * 16 + (lane & 15)) * BPITCH;
#pragma unroll
      for (int v = 0; v < 4; ++v) {
        fb[nt].u[v]     = *(const unsigned*)(brow + kb + 2 * v);
        fb[nt].u[4 + v] = *(const unsigned*)(brow + 16 + kb + 2 * v);
      }
    }
#pragma unroll
    for (int nt = 0; nt < 4; ++nt)
      acc[nt] = __builtin_amdgcn_wmma_f32_16x16x32_bf16(false, fa.v, false, fb[nt].v,
                                                        (short)0, acc[nt], false, false);

    // Scatter next B tile (transposed) after the WMMAs so the loadcnt wait
    // overlaps with compute.
    if (have_next) {
#pragma unroll
      for (int e = 0; e < 8; ++e) Bs[nxt][(bc + e) * BPITCH + br] = ub.s[e];
    }
    wait_async0();
    __syncthreads();
  }

  // C/D layout: VGPR r -> row = r + (lane<16 ? 0 : 8), col = lane&15
  const int mbase = m0 + wave * 16 + ((lane >> 4) << 3);
  const int nbl   = n0 + (lane & 15);
#pragma unroll
  for (int r = 0; r < 8; ++r) {
    int m = mbase + r;
    float bv = bias ? bias[m] : 0.0f;
    float g  = (epi == EPI_RES) ? gamma[m] : 0.0f;
#pragma unroll
    for (int nt = 0; nt < 4; ++nt) {
      float x = acc[nt][r] + bv;
      if (epi == EPI_GELU) x = 0.5f * x * (1.0f + erff(x * 0.70710678118654752f));
      size_t o = (size_t)m * ldc + (size_t)(nbl + nt * 16) * nstride;
      if (epi == EPI_RES) x = resid[o] + g * x;
      C[o] = x;
    }
  }
}

// ---------------------------------------------------------------------------
// Depthwise 7-tap conv (pad 3), act layout (512, T)
// ---------------------------------------------------------------------------
__global__ void k_dwconv7(const float* __restrict__ in, const float* __restrict__ w,
                          const float* __restrict__ bias, float* __restrict__ out, int T) {
  int c = blockIdx.y;
  int t = blockIdx.x * 256 + threadIdx.x;
  if (t >= T) return;
  const float* row = in + (size_t)c * T;
  const float* wc = w + c * 7;
  float acc = bias[c];
#pragma unroll
  for (int j = 0; j < 7; ++j) {
    int tt = t + j - 3;
    if (tt >= 0 && tt < T) acc += row[tt] * wc[j];
  }
  out[(size_t)c * T + t] = acc;
}

// LayerNorm over channel (512) per column t, in-place, layout (512, T)
__global__ void k_layernorm(float* __restrict__ y, const float* __restrict__ g,
                            const float* __restrict__ b, int T) {
  int t = blockIdx.x * 256 + threadIdx.x;
  if (t >= T) return;
  float s = 0.f;
  for (int c = 0; c < 512; ++c) s += y[(size_t)c * T + t];
  float mu = s * (1.0f / 512.0f);
  float v = 0.f;
  for (int c = 0; c < 512; ++c) { float d = y[(size_t)c * T + t] - mu; v += d * d; }
  float rs = rsqrtf(v * (1.0f / 512.0f) + 1e-6f);
  for (int c = 0; c < 512; ++c) {
    size_t o = (size_t)c * T + t;
    y[o] = (y[o] - mu) * rs * g[c] + b[c];
  }
}

// VQ argmin: d(j,t) = ||cb_j||^2 - 2*S[j,t] (+||z||^2 const per t); first-min ties
__global__ void k_vq_argmin(const float* __restrict__ S, const float* __restrict__ n2,
                            int* __restrict__ idx, int N) {
  int t = blockIdx.x * 256 + threadIdx.x;
  if (t >= N) return;
  float best = 3.4e38f;
  int bi = 0;
  for (int j = 0; j < 1024; ++j) {
    float d = n2[j] - 2.0f * S[(size_t)j * N + t];
    if (d < best) { best = d; bi = j; }
  }
  idx[t] = bi;
}

// q[c][t] = cb[idx[t]][c]  (straight-through forward value)
__global__ void k_gather_q(const int* __restrict__ idx, const float* __restrict__ cb,
                           float* __restrict__ q, int T) {
  int i = blockIdx.x * 256 + threadIdx.x;
  if (i >= 512 * T) return;
  int c = i / T, t = i - c * T;
  q[i] = cb[(size_t)idx[t] * 512 + c];
}

__global__ void k_commit(const float* __restrict__ q, const float* __restrict__ z,
                         float* __restrict__ loss, int n, float scale) {
  __shared__ float sm[256];
  int i = blockIdx.x * 256 + threadIdx.x;
  float d = 0.f;
  if (i < n) { float e = q[i] - z[i]; d = e * e * scale; }
  sm[threadIdx.x] = d;
  __syncthreads();
  for (int s = 128; s > 0; s >>= 1) {
    if (threadIdx.x < s) sm[threadIdx.x] += sm[threadIdx.x + s];
    __syncthreads();
  }
  if (threadIdx.x == 0) atomicAdd(loss, sm[0]);
}

// ---------------------------------------------------------------------------
// Host orchestration
// ---------------------------------------------------------------------------
extern "C" void kernel_launch(void* const* d_in, const int* in_sizes, int n_in,
                              void* d_out, int out_size, void* d_ws, size_t ws_size,
                              hipStream_t stream) {
  (void)in_sizes; (void)n_in; (void)out_size; (void)ws_size;
  auto P = [&](int i) { return (const float*)d_in[i]; };
  // Input order: x, [down0: conv_w,conv_b,dw_w,dw_b,ln_g,ln_b,w1,b1,w2,b2,gamma],
  //              [down1: ...], codebook, [up0: ...], [up1: ...]
  const int sb[4] = {1, 12, 24, 35};  // stage param bases: down0, down1, up0, up1

  char* W = (char*)d_ws;
  size_t off = 0;
  auto alloc = [&](size_t bytes) -> void* {
    off = (off + 255) & ~(size_t)255;
    void* p = W + off;
    off += bytes;
    return p;
  };

  bf16_t* wAd[2];   wAd[0] = (bf16_t*)alloc(524288 * 2); wAd[1] = (bf16_t*)alloc(524288 * 2);
  bf16_t* wAu[2][2];
  for (int s = 0; s < 2; ++s)
    for (int k = 0; k < 2; ++k) wAu[s][k] = (bf16_t*)alloc(262144 * 2);
  bf16_t* w1b[4]; bf16_t* w2b[4];
  for (int s = 0; s < 4; ++s) {
    w1b[s] = (bf16_t*)alloc((size_t)2048 * 512 * 2);
    w2b[s] = (bf16_t*)alloc((size_t)512 * 2048 * 2);
  }
  bf16_t* cbb  = (bf16_t*)alloc(524288 * 2);
  float*  cbn2 = (float*)alloc(1024 * 4);
  float*  h1 = (float*)alloc((size_t)16 * 512 * 2048 * 4);
  float*  h2 = (float*)alloc((size_t)16 * 512 * 1024 * 4);
  float*  qb = (float*)alloc((size_t)16 * 512 * 1024 * 4);
  float*  u1 = (float*)alloc((size_t)16 * 512 * 2048 * 4);
  float*  dwbuf = (float*)alloc((size_t)512 * 4096 * 4);
  bf16_t* actb  = (bf16_t*)alloc((size_t)2048 * 4096 * 2);
  float*  mlp   = (float*)alloc((size_t)2048 * 4096 * 4);
  bf16_t* convB = (bf16_t*)alloc((size_t)1024 * 2048 * 2);
  float*  Smat  = (float*)alloc((size_t)1024 * 1024 * 4);

  float* outF = (float*)d_out;
  int*   outI = (int*)(outF + (size_t)16 * 512 * 4096);
  float* lossP = outF + (size_t)16 * 512 * 4096 + 16384;

  auto cvt = [&](const float* s, bf16_t* d, int n) {
    k_f32_to_bf16<<<dim3((n + 255) / 256), 256, 0, stream>>>(s, d, n);
  };
  auto gemm = [&](const bf16_t* A, const bf16_t* B, float* C, int M, int N, int K,
                  int ldc, int ns, const float* bias, const float* gamma,
                  const float* resid, int epi) {
    k_gemm_wmma<<<dim3(N / 64, M / 128), 256, 0, stream>>>(A, B, C, N, K, ldc, ns,
                                                           bias, gamma, resid, epi);
  };
  auto convnext = [&](float* act, int T, int s) {
    int base = sb[s];
    k_dwconv7<<<dim3((T + 255) / 256, 512), 256, 0, stream>>>(act, P(base + 2),
                                                              P(base + 3), dwbuf, T);
    k_layernorm<<<dim3((T + 255) / 256), 256, 0, stream>>>(dwbuf, P(base + 4),
                                                           P(base + 5), T);
    cvt(dwbuf, actb, 512 * T);
    gemm(w1b[s], actb, mlp, 2048, T, 512, T, 1, P(base + 7), nullptr, nullptr, EPI_GELU);
    cvt(mlp, actb, 2048 * T);
    gemm(w2b[s], actb, act, 512, T, 2048, T, 1, P(base + 9), P(base + 10), act, EPI_RES);
  };

  // ---- one-time weight packing ----
  cvt(P(sb[0]), wAd[0], 524288);
  cvt(P(sb[1]), wAd[1], 524288);
  k_pack_upconv_A<<<dim3(1024), 256, 0, stream>>>(P(sb[2]), wAu[0][0], wAu[0][1]);
  k_pack_upconv_A<<<dim3(1024), 256, 0, stream>>>(P(sb[3]), wAu[1][0], wAu[1][1]);
  for (int s = 0; s < 4; ++s) {
    cvt(P(sb[s] + 6), w1b[s], 2048 * 512);
    cvt(P(sb[s] + 8), w2b[s], 512 * 2048);
  }
  cvt(P(23), cbb, 524288);
  k_cb_norms<<<4, 256, 0, stream>>>(P(23), cbn2);
  k_zero<<<1, 1, 0, stream>>>(lossP);

  const float* x = P(0);
  for (int b = 0; b < 16; ++b) {
    const float* xb = x + (size_t)b * 512 * 4096;
    float* h1b = h1 + (size_t)b * 512 * 2048;
    float* h2b = h2 + (size_t)b * 512 * 1024;
    float* qbb = qb + (size_t)b * 512 * 1024;
    float* u1b = u1 + (size_t)b * 512 * 2048;
    float* ob  = outF + (size_t)b * 512 * 4096;

    // down stage 0: (512,4096) -> (512,2048)
    k_pack_downconv_B<<<dim3(1024 * 2048 / 256), 256, 0, stream>>>(xb, convB, 4096);
    gemm(wAd[0], convB, h1b, 512, 2048, 1024, 2048, 1, P(sb[0] + 1), nullptr, nullptr, EPI_BIAS);
    convnext(h1b, 2048, 0);

    // down stage 1: (512,2048) -> (512,1024)
    k_pack_downconv_B<<<dim3(1024 * 1024 / 256), 256, 0, stream>>>(h1b, convB, 2048);
    gemm(wAd[1], convB, h2b, 512, 1024, 1024, 1024, 1, P(sb[1] + 1), nullptr, nullptr, EPI_BIAS);
    convnext(h2b, 1024, 1);

    // VQ: scores S[j,t] = cb_j . z_t via WMMA, then argmin/gather/loss
    cvt(h2b, actb, 512 * 1024);
    gemm(cbb, actb, Smat, 1024, 1024, 512, 1024, 1, nullptr, nullptr, nullptr, EPI_BIAS);
    k_vq_argmin<<<4, 256, 0, stream>>>(Smat, cbn2, outI + b * 1024, 1024);
    k_gather_q<<<dim3(512 * 1024 / 256), 256, 0, stream>>>(outI + b * 1024, P(23), qbb, 1024);
    k_commit<<<dim3(512 * 1024 / 256), 256, 0, stream>>>(qbb, h2b, lossP, 512 * 1024,
                                                         1.0f / (16.0f * 1024.0f * 512.0f));

    // up stage 0: (512,1024) -> (512,2048), two interleaved GEMMs
    cvt(qbb, actb, 512 * 1024);
    gemm(wAu[0][0], actb, u1b + 0, 512, 1024, 512, 2048, 2, P(sb[2] + 1), nullptr, nullptr, EPI_BIAS);
    gemm(wAu[0][1], actb, u1b + 1, 512, 1024, 512, 2048, 2, P(sb[2] + 1), nullptr, nullptr, EPI_BIAS);
    convnext(u1b, 2048, 2);

    // up stage 1: (512,2048) -> (512,4096), write directly into d_out h region
    cvt(u1b, actb, 512 * 2048);
    gemm(wAu[1][0], actb, ob + 0, 512, 2048, 512, 4096, 2, P(sb[3] + 1), nullptr, nullptr, EPI_BIAS);
    gemm(wAu[1][1], actb, ob + 1, 512, 2048, 512, 4096, 2, P(sb[3] + 1), nullptr, nullptr, EPI_BIAS);
    convnext(ob, 4096, 3);
  }
}
